// Monotonic_attention_train_10797547782312
// MI455X (gfx1250) — matrix operand order repaired and checked by reading, not compiled, not run on hardware
//
#include <hip/hip_runtime.h>
#include <hip/hip_bf16.h>
#include <math.h>

// ---------------- dims ----------------
#define H   512
#define H2  1024
#define H4  2048
#define C10 10000
#define T   2048
#define WIN 4
#define MAXSTEP 80

// ---------------- state ints ----------------
enum { ST_PREV = 0, ST_HAS, ST_CNT, ST_DONE, ST_FI, ST_ACT, ST_TOK };

typedef __attribute__((ext_vector_type(16))) _Float16 v16h;
typedef __attribute__((ext_vector_type(8)))  float    v8f;

__device__ __forceinline__ float sigf(float x) { return 1.0f / (1.0f + __expf(-x)); }

// packed argmax key: max value wins; equal value -> smaller index wins
__device__ __forceinline__ unsigned long long packKey(float v, unsigned idx) {
    unsigned u = __float_as_uint(v);
    u = (u & 0x80000000u) ? ~u : (u | 0x80000000u);   // order-preserving f32->u32
    return ((unsigned long long)u << 32) | (unsigned long long)(0xFFFFFFFFu - idx);
}

// ============================================================
// Setup: weight transpose + f32->f16 convert  Wt[n*K+k] = W[k*N+n]
// ============================================================
__global__ void k_convT(const float* __restrict__ W, _Float16* __restrict__ Wt) {
    int idx = blockIdx.x * blockDim.x + threadIdx.x;      // 1M elements
    int n = idx >> 10, k = idx & 1023;
    Wt[idx] = (_Float16)W[k * H2 + n];
}

// ============================================================
// WMMA GEMM: D[T,H2] = A[T,K=1024](f32) x B (Bt = B^T [N,K] f16)
// one wave -> one 16x16 tile, block = 8 waves = 16x128 tile
// ============================================================
__global__ __launch_bounds__(256) void k_gemm_wmma(
    const float* __restrict__ A, const _Float16* __restrict__ Bt,
    float* __restrict__ D) {
    const int K = H2, N = H2;
    int wave = threadIdx.x >> 5;
    int lane = threadIdx.x & 31;
    int n16  = lane & 15;
    int hi   = lane >> 4;                 // 0: lanes 0-15, 1: lanes 16-31
    int tm = blockIdx.x * 16;             // 128 blocks
    int tn = blockIdx.y * 128 + wave * 16;// 8 blocks
    const float*    arow = A  + (size_t)(tm + n16) * K;   // A row for M=lane%16
    const _Float16* brow = Bt + (size_t)(tn + n16) * K;   // B column (transposed row)
    const int ak = hi * 16;               // A K-base per half-wave
    const int bk = hi * 16;               // B K-base per half-wave
    v8f acc = {};
    for (int kk = 0; kk < K; kk += 32) {
        v16h a, b;
#pragma unroll
        for (int u = 0; u < 8; ++u) {
            a[u]     = (_Float16)arow[kk + ak + u];        // K = ak..ak+7
            a[u + 8] = (_Float16)arow[kk + ak + 8 + u];    // K = ak+8..ak+15
        }
        b = *(const v16h*)(brow + kk + bk);                // K = bk..bk+15 contiguous
        acc = __builtin_amdgcn_wmma_f32_16x16x32_f16(
            false, a, false, b, (short)0, acc, false, false);
    }
#pragma unroll
    for (int i = 0; i < 8; ++i)
        D[(size_t)(tm + i + hi * 8) * N + tn + n16] = acc[i];
}

// ============================================================
// init: vnorm = g/||vm||, zero s,c, reset decode state + argmax key
// ============================================================
__global__ void k_init(const float* __restrict__ vm, const float* __restrict__ g,
                       float* s, float* c, float* vnp, int* st,
                       unsigned long long* key) {
    __shared__ float red[512];
    float p = 0.f;
    for (int j = threadIdx.x; j < H2; j += 512) p += vm[j] * vm[j];
    red[threadIdx.x] = p; __syncthreads();
    for (int off = 256; off > 0; off >>= 1) {
        if (threadIdx.x < off) red[threadIdx.x] += red[threadIdx.x + off];
        __syncthreads();
    }
    if (threadIdx.x == 0) {
        vnp[0] = g[0] / sqrtf(red[0]);
        st[ST_PREV] = 0; st[ST_HAS] = 1; st[ST_CNT] = 0; st[ST_DONE] = 0;
        st[ST_FI] = 0; st[ST_ACT] = 0; st[ST_TOK] = 0;
        key[0] = 0ull;
    }
    s[threadIdx.x] = 0.f; c[threadIdx.x] = 0.f;
}

// ============================================================
// u_m = s@Wsm + b ; u_c = s@Wsc + b     (grid 8 x 256)
// ============================================================
__global__ void k_proj(const float* __restrict__ s,
                       const float* __restrict__ Wsm, const float* __restrict__ Wsmb,
                       const float* __restrict__ Wsc, const float* __restrict__ Wscb,
                       float* __restrict__ u_m, float* __restrict__ u_c,
                       const int* __restrict__ st) {
    if (st[ST_DONE]) return;              // done latches; outputs frozen
    int b = blockIdx.x;
    const float *W, *bias; float* out; int j;
    if (b < 4) { W = Wsm; bias = Wsmb; out = u_m; j = b * 256 + threadIdx.x; }
    else       { W = Wsc; bias = Wscb; out = u_c; j = (b - 4) * 256 + threadIdx.x; }
    float acc = bias[j];
    for (int i = 0; i < H; ++i) acc += s[i] * W[i * H2 + j];
    out[j] = acc;
}

// ============================================================
// e_mono[t] = vnorm * sum_j tanh(encM[t,j]+u_m[j]) * vm[j] + r   (grid T x 256)
// only frames t >= prev_idx can fire (cumsum of one-hot mask)
// ============================================================
__global__ void k_energy(const float* __restrict__ encM, const float* __restrict__ u_m,
                         const float* __restrict__ vm, const float* __restrict__ vnp,
                         const float* __restrict__ r_mono, float* __restrict__ e,
                         const int* __restrict__ st) {
    int t = blockIdx.x;
    if (st[ST_DONE] || t < st[ST_PREV]) return;
    float p = 0.f;
    for (int j = threadIdx.x; j < H2; j += 256)
        p += tanhf(encM[(size_t)t * H2 + j] + u_m[j]) * vm[j];
    __shared__ float red[256];
    red[threadIdx.x] = p; __syncthreads();
    for (int off = 128; off > 0; off >>= 1) {
        if (threadIdx.x < off) red[threadIdx.x] += red[threadIdx.x + off];
        __syncthreads();
    }
    if (threadIdx.x == 0) e[t] = vnp[0] * red[0] + r_mono[0];
}

// ============================================================
// fired-frame scan + state machine + chunkwise attention  (1 block x 256)
// sigmoid(e)>=0.5 <=> e>=0 ; invalid window entries pad win_e with 0.0
// ============================================================
__global__ void k_selchunk(const float* __restrict__ e, int* st,
                           const float* __restrict__ encC, const float* __restrict__ u_c,
                           const float* __restrict__ vc, const float* __restrict__ enc,
                           float* __restrict__ ctx) {
    if (st[ST_DONE]) return;
    __shared__ int sMin, sFi, sAct;
    if (threadIdx.x == 0) sMin = T;
    __syncthreads();
    int prev = st[ST_PREV], hasAtt = st[ST_HAS];
    int best = T;
    if (hasAtt) {
        for (int t = threadIdx.x; t < T; t += 256)
            if (t >= prev && e[t] >= 0.f) { best = t; break; }
    }
    atomicMin(&sMin, best);
    __syncthreads();
    if (threadIdx.x == 0) {
        int fired  = (sMin < T);
        int fi     = fired ? sMin : 0;
        int same   = fired && hasAtt && (fi == prev);
        int cs     = same ? (st[ST_CNT] + 1) : 0;
        int fired2 = fired && (cs != 2);
        int done   = st[ST_DONE];
        int active = (!done) && fired2;
        st[ST_CNT] = cs;
        st[ST_DONE] = done || (!fired2);
        st[ST_FI] = fi;
        st[ST_ACT] = active;
        if (active) { st[ST_PREV] = fi; st[ST_HAS] = 1; }
        sFi = fi; sAct = active;
    }
    __syncthreads();
    if (!sAct) return;                     // ctx unused when inactive
    // ---- chunkwise attention over window ending at fi ----
    __shared__ float red[256];
    __shared__ float beta[WIN];
    __shared__ int sIdx[WIN], sValid[WIN];
    int fi = sFi;
    int start = fi + 1 - WIN; if (start < 0) start = 0;
    for (int w = 0; w < WIN; ++w) {
        int idx = start + w;
        int valid = (idx <= fi);
        float p = 0.f;
        for (int j = threadIdx.x; j < H2; j += 256)
            p += tanhf(encC[(size_t)idx * H2 + j] + u_c[j]) * vc[j];
        red[threadIdx.x] = p; __syncthreads();
        for (int off = 128; off > 0; off >>= 1) {
            if (threadIdx.x < off) red[threadIdx.x] += red[threadIdx.x + off];
            __syncthreads();
        }
        if (threadIdx.x == 0) { beta[w] = valid ? red[0] : 0.f; sIdx[w] = idx; sValid[w] = valid; }
        __syncthreads();
    }
    if (threadIdx.x == 0) {
        float mx = beta[0];
        for (int w = 1; w < WIN; ++w) mx = fmaxf(mx, beta[w]);
        float ex[WIN], sum = 0.f;
        for (int w = 0; w < WIN; ++w) { ex[w] = __expf(beta[w] - mx); sum += ex[w]; }
        for (int w = 0; w < WIN; ++w) beta[w] = ex[w] / sum;
    }
    __syncthreads();
    for (int j = threadIdx.x; j < H2; j += 256) {
        float v = 0.f;
        for (int w = 0; w < WIN; ++w)
            if (sValid[w]) v += beta[w] * enc[(size_t)sIdx[w] * H2 + j];
        ctx[j] = v;
    }
}

// ============================================================
// h = tanh(ctx@Lgy + Lgy_b + s@Lsy)    (grid 2 x 256)
// ============================================================
__global__ void k_hidden(const float* __restrict__ ctx, const float* __restrict__ s,
                         const float* __restrict__ Lgy, const float* __restrict__ Lgyb,
                         const float* __restrict__ Lsy, float* __restrict__ h,
                         const int* __restrict__ st) {
    if (!st[ST_ACT]) return;
    int j = blockIdx.x * 256 + threadIdx.x;     // < 512
    float acc = Lgyb[j];
    for (int i = 0; i < H2; ++i) acc += ctx[i] * Lgy[i * H + j];
    for (int i = 0; i < H;  ++i) acc += s[i]   * Lsy[i * H + j];
    h[j] = tanhf(acc);
}

// ============================================================
// logits: y = h@Lyy + b ; out = active?y:0 ; fused atomic argmax  (grid 40 x 256)
// ============================================================
__global__ void k_logits(const float* __restrict__ h, const float* __restrict__ Lyy,
                         const float* __restrict__ Lyyb, const int* __restrict__ st,
                         float* __restrict__ out_row, unsigned long long* __restrict__ key) {
    int j = blockIdx.x * 256 + threadIdx.x;
    if (j >= C10) return;
    if (!st[ST_ACT]) { out_row[j] = 0.f; return; }   // must still emit zero row
    float acc = Lyyb[j];
    for (int i = 0; i < H; ++i) acc += h[i] * Lyy[(size_t)i * C10 + j];
    out_row[j] = acc;
    atomicMax(key, packKey(acc, (unsigned)j));
}

// ============================================================
// rec = emb[tok] + s@Lss + ctx@Lgs + Lgs_b     (grid 8 x 256)
// ============================================================
__global__ void k_rec(const float* __restrict__ s, const float* __restrict__ ctx,
                      const float* __restrict__ Lss, const float* __restrict__ Lgs,
                      const float* __restrict__ Lgsb, const float* __restrict__ emb,
                      const int* __restrict__ st, const unsigned long long* __restrict__ key,
                      float* __restrict__ rec) {
    if (!st[ST_ACT]) return;
    int j = blockIdx.x * 256 + threadIdx.x;     // < 2048
    int tok = (int)(0xFFFFFFFFu - (unsigned)(key[0] & 0xFFFFFFFFull));
    float acc = emb[(size_t)tok * H4 + j] + Lgsb[j];
    for (int i = 0; i < H;  ++i) acc += s[i]   * Lss[(size_t)i * H4 + j];
    for (int i = 0; i < H2; ++i) acc += ctx[i] * Lgs[(size_t)i * H4 + j];
    rec[j] = acc;
}

// ============================================================
// gated LSTM update + argmax-key reset      (1 block x 512)
// ============================================================
__global__ void k_update(const float* __restrict__ rec, const int* __restrict__ st,
                         unsigned long long* __restrict__ key, float* s, float* c) {
    if (!st[ST_ACT]) return;
    int k = threadIdx.x;
    int tok = (int)(0xFFFFFFFFu - (unsigned)(key[0] & 0xFFFFFFFFull));
    float ig = rec[k], fg = rec[H + k], gg = rec[2 * H + k], og = rec[3 * H + k];
    float cn = sigf(fg) * c[k] + sigf(ig) * tanhf(gg);
    float sn = sigf(og) * tanhf(cn);
    if (tok == 1) { sn = 0.f; cn = 0.f; }
    s[k] = sn; c[k] = cn;
    __syncthreads();                       // all lanes consumed key
    if (k == 0) key[0] = 0ull;             // reset for next step's argmax
}

// ============================================================
extern "C" void kernel_launch(void* const* d_in, const int* in_sizes, int n_in,
                              void* d_out, int out_size, void* d_ws, size_t ws_size,
                              hipStream_t stream) {
    const float* enc   = (const float*)d_in[0];   // [T,H2]
    const float* Wsm   = (const float*)d_in[1];
    const float* Wsmb  = (const float*)d_in[2];
    const float* Whm   = (const float*)d_in[3];
    const float* vm    = (const float*)d_in[4];
    const float* gmono = (const float*)d_in[5];
    const float* rmono = (const float*)d_in[6];
    const float* Wsc   = (const float*)d_in[7];
    const float* Wscb  = (const float*)d_in[8];
    const float* Whc   = (const float*)d_in[9];
    const float* vc    = (const float*)d_in[10];
    const float* Lsy   = (const float*)d_in[11];
    const float* Lgy   = (const float*)d_in[12];
    const float* Lgyb  = (const float*)d_in[13];
    const float* Lyy   = (const float*)d_in[14];
    const float* Lyyb  = (const float*)d_in[15];
    const float* Lemb  = (const float*)d_in[16];
    const float* Lss   = (const float*)d_in[17];
    const float* Lgs   = (const float*)d_in[18];
    const float* Lgsb  = (const float*)d_in[19];
    float* out = (float*)d_out;                   // [80, 10000]

    char* ws = (char*)d_ws;
    _Float16* WT_M  = (_Float16*)(ws);                       // 2 MB
    _Float16* WT_C  = (_Float16*)(ws + (2u  << 20));         // 2 MB
    float*    ENC_M = (float*)   (ws + (4u  << 20));         // 8 MB
    float*    ENC_C = (float*)   (ws + (12u << 20));         // 8 MB
    float*    SC    = (float*)   (ws + (20u << 20));
    float* u_m  = SC;            // 1024
    float* u_c  = SC + 1024;     // 1024
    float* e_m  = SC + 2048;     // 2048
    float* ctx  = SC + 4096;     // 1024
    float* hbuf = SC + 5120;     // 512
    float* rec  = SC + 5632;     // 2048
    float* sv   = SC + 7680;     // 512
    float* cv   = SC + 8192;     // 512
    float* vnp  = SC + 8704;     // 1
    int*   st   = (int*)(SC + 8712);                    // 8 ints
    unsigned long long* key = (unsigned long long*)(SC + 8720); // 8B aligned (byte off %8==0)

    // ---- one-time per call: convert weights, WMMA encoder projections, init ----
    k_convT<<<4096, 256, 0, stream>>>(Whm, WT_M);
    k_convT<<<4096, 256, 0, stream>>>(Whc, WT_C);
    dim3 ggrid(T / 16, H2 / 128);
    k_gemm_wmma<<<ggrid, 256, 0, stream>>>(enc, WT_M, ENC_M);
    k_gemm_wmma<<<ggrid, 256, 0, stream>>>(enc, WT_C, ENC_C);
    k_init<<<1, 512, 0, stream>>>(vm, gmono, sv, cv, vnp, st, key);

    // ---- 80 sequential decode steps (7 kernels/step) ----
    for (int step = 0; step < MAXSTEP; ++step) {
        k_proj    <<<8, 256, 0, stream>>>(sv, Wsm, Wsmb, Wsc, Wscb, u_m, u_c, st);
        k_energy  <<<T, 256, 0, stream>>>(ENC_M, u_m, vm, vnp, rmono, e_m, st);
        k_selchunk<<<1, 256, 0, stream>>>(e_m, st, ENC_C, u_c, vc, enc, ctx);
        k_hidden  <<<2, 256, 0, stream>>>(ctx, sv, Lgy, Lgyb, Lsy, hbuf, st);
        k_logits  <<<40, 256, 0, stream>>>(hbuf, Lyy, Lyyb, st,
                                           out + (size_t)step * C10, key);
        k_rec     <<<8, 256, 0, stream>>>(sv, ctx, Lss, Lgs, Lgsb, Lemb, st, key, rec);
        k_update  <<<1, 512, 0, stream>>>(rec, st, key, sv, cv);
    }
}
